// ProbAttention_old_71631464563437
// MI455X (gfx1250) — matrix operand order repaired and checked by reading, not compiled, not run on hardware
//
#include <hip/hip_runtime.h>
#include <hip/hip_bf16.h>
#include <stddef.h>
#include <stdint.h>

// ---------------------------------------------------------------------------
// ProbSparse attention (Informer) for MI455X / gfx1250, wave32.
// WMMA f16 GEMMs + TDM (tensor_load_to_lds) double-stage K/V pipeline.
// Shapes fixed by the reference: B=4, L=2048, H=8, D=64, sample_k=n_top=40.
// ---------------------------------------------------------------------------

#define L_Q   2048
#define DIM   64
#define NHEAD 8
#define NBAT  4
#define SAMPK 40
#define NTOP  40
#define MPAD  48      // 40 rounded up to 3 WMMA M-tiles
#define KBLK  64      // keys per streaming block
#define NKB   (L_Q / KBLK)
#define VSTR  72      // padded LDS stride (halves) for transposed V
#define NCH   8       // cumsum chunks
#define CHSZ  (L_Q / NCH)
#define NEGF  (-3.0e38f)

typedef _Float16 f16;
typedef __attribute__((ext_vector_type(16))) _Float16     v16h;
typedef __attribute__((ext_vector_type(8)))  float        v8f;
typedef __attribute__((ext_vector_type(4)))  unsigned int u32x4;
typedef __attribute__((ext_vector_type(8)))  int          i32x8;
typedef __attribute__((ext_vector_type(4)))  int          i32x4;

static __device__ inline v8f zero_v8f() {
  v8f z;
#pragma unroll
  for (int i = 0; i < 8; ++i) z[i] = 0.0f;
  return z;
}

// A-operand (16x32 f16, M x K) loaded from row-major LDS [row][stride halves].
// Layout: lane = {hs=lane>>4, M=lane&15};
// VGPR j -> K = hs*16 + 2j.. (contiguous, matches load_Bt16)
static __device__ inline v16h load_A16(const f16* __restrict__ base, int stride) {
  const int lane = threadIdx.x & 31;
  const int M  = lane & 15;
  const int hs = lane >> 4;
  const f16* row = base + M * stride;
  v16h a;
#pragma unroll
  for (int j = 0; j < 8; ++j) {
    int kk = hs * 16 + 2 * j;
    a[2 * j]     = row[kk];
    a[2 * j + 1] = row[kk + 1];
  }
  return a;
}

// B-operand (32x16 f16, K x N) with B[k][n] = storage[n][k] (storage row = n).
// Lanes 0-15 hold K=0..15, lanes 16-31 hold K=16..31; 16 contiguous halves.
static __device__ inline v16h load_Bt16(const f16* __restrict__ base, int stride) {
  const int lane = threadIdx.x & 31;
  const int N   = lane & 15;
  const int grp = lane >> 4;
  const f16* p = base + N * stride + grp * 16;
  v16h b;
#pragma unroll
  for (int hh = 0; hh < 16; ++hh) b[hh] = p[hh];
  return b;
}

static __device__ inline v8f wmma_f16(v16h a, v16h b, v8f c) {
  return __builtin_amdgcn_wmma_f32_16x16x32_f16(
      /*neg_a=*/false, a, /*neg_b=*/false, b,
      /*c_mod=*/(short)0, c, /*reuse_a=*/false, /*reuse_b=*/false);
}

// ---------------------------------------------------------------------------
// TDM: DMA one 64x64 f32 tile (rows stride H*D elements, i.e. the (B,L,H,D)
// layout at fixed b,h) from global into LDS. Descriptor per CDNA5 ISA
// §8.3-8.5: 2D tile, groups 2/3 zero. Issued by one wave; tracked on TENSORcnt.
// This toolchain's builtin is the 6-arg clang-23 form:
//   (u32x4 g0, i32x8 g1, i32x4 g2, i32x4 g3, i32x8, i32 cpol)
// ---------------------------------------------------------------------------
static __device__ inline void tdm_load_tile_64x64_f32(const float* gsrc, void* lds_dst) {
  const unsigned long long ga = (unsigned long long)(uintptr_t)gsrc;
  const unsigned int laddr = (unsigned int)(uintptr_t)lds_dst; // low 32b = LDS offset

  u32x4 g0;
  g0[0] = 1u;                                            // count=1 (valid), user mode
  g0[1] = laddr;                                         // lds_addr [63:32]
  g0[2] = (unsigned int)(ga & 0xffffffffu);              // global_addr lo
  g0[3] = (unsigned int)((ga >> 32) & 0x01ffffffu)       // global_addr[56:32]
        | (2u << 30);                                    // type=2 ("image")

  i32x8 g1;
  g1[0] = 0x00020000;                                    // data_size=2 (4B); mask=0
  g1[1] = (int)((DIM & 0xffff) << 16);                   // tensor_dim0 lo16 @ [63:48]
  g1[2] = (int)((DIM >> 16) | ((KBLK & 0xffff) << 16));  // tdim0 hi | tensor_dim1 lo
  g1[3] = (int)((KBLK >> 16) | (DIM << 16));             // tdim1 hi | tile_dim0=64
  g1[4] = (int)KBLK;                                     // tile_dim1=64, tile_dim2=0
  g1[5] = (int)(NHEAD * DIM);                            // tensor_dim0_stride = 512
  g1[6] = 0;                                             // stride hi | tdim1_stride lo
  g1[7] = 0;

  const i32x4 z4 = {0, 0, 0, 0};
  const i32x8 z8 = {0, 0, 0, 0, 0, 0, 0, 0};
  __builtin_amdgcn_tensor_load_to_lds(g0, g1, z4, z4, z8, 0);
}

// ---------------------------------------------------------------------------
// Kernel 1: sampling sparsity measure  M[b,h,l] = max_s(q.k_s) - mean_s over L
// One wave32 per query row; lanes hold float2 of the D=64 Q row.
// ---------------------------------------------------------------------------
__global__ __launch_bounds__(256) void sample_score_kernel(
    const float* __restrict__ qg, const float* __restrict__ kg,
    const long long* __restrict__ idxs, float* __restrict__ Mout) {
  const int wid  = (int)((blockIdx.x * blockDim.x + threadIdx.x) >> 5);
  const int lane = threadIdx.x & 31;
  const int total = NBAT * NHEAD * L_Q;
  if (wid >= total) return;
  const int l  = wid % L_Q;
  const int bh = wid / L_Q;
  const int b  = bh / NHEAD;
  const int h  = bh % NHEAD;

  const float* qrow = qg + (((size_t)b * L_Q + l) * NHEAD + h) * DIM;
  const float2 qv = *(const float2*)(qrow + lane * 2);

  float mx = NEGF, sm = 0.0f;
#pragma unroll 4
  for (int s = 0; s < SAMPK; ++s) {
    const int ki = (int)idxs[(size_t)l * SAMPK + s];
    const float* krow = kg + (((size_t)b * L_Q + ki) * NHEAD + h) * DIM;
    const float2 kv = *(const float2*)(krow + lane * 2);
    float p = qv.x * kv.x + qv.y * kv.y;
#pragma unroll
    for (int off = 16; off >= 1; off >>= 1) p += __shfl_xor(p, off, 32);
    mx = fmaxf(mx, p);
    sm += p;
  }
  if (lane == 0) Mout[(size_t)bh * L_Q + l] = mx - sm / (float)L_Q;
}

// ---------------------------------------------------------------------------
// Kernel 2: top-40 indices of M per (b,h). Iterative argmax (matches top_k
// ordering; ties break to the smaller index). One block per (b,h).
// ---------------------------------------------------------------------------
__global__ __launch_bounds__(256) void topk_kernel(
    const float* __restrict__ Min, int* __restrict__ mtop) {
  __shared__ float vals[L_Q];
  __shared__ float sv[256];
  __shared__ int   si[256];
  const int bh = blockIdx.x, tid = threadIdx.x;

  for (int i = tid; i < L_Q; i += 256) vals[i] = Min[(size_t)bh * L_Q + i];
  __syncthreads();

  for (int t = 0; t < NTOP; ++t) {
    float bv = NEGF;
    int bi = 0x7fffffff;
    for (int i = tid; i < L_Q; i += 256) {
      const float vv = vals[i];
      if (vv > bv || (vv == bv && i < bi)) { bv = vv; bi = i; }
    }
    sv[tid] = bv; si[tid] = bi;
    __syncthreads();
    for (int st = 128; st >= 1; st >>= 1) {
      if (tid < st) {
        const float ov = sv[tid + st];
        const int   oi = si[tid + st];
        if (ov > sv[tid] || (ov == sv[tid] && oi < si[tid])) { sv[tid] = ov; si[tid] = oi; }
      }
      __syncthreads();
    }
    if (tid == 0) { mtop[bh * NTOP + t] = si[0]; vals[si[0]] = NEGF; }
    __syncthreads();
  }
}

// ---------------------------------------------------------------------------
// Kernels 3a/3b: context = cumsum(V, axis=L) in (B,H,L,D) layout.
// Chunked scan: per-(b,h,chunk) block of 64 threads (one lane per d),
// coalesced 256B rows; then a prefix fix-up pass.
// ---------------------------------------------------------------------------
__global__ __launch_bounds__(64) void cumsum_part_kernel(
    const float* __restrict__ vg, float* __restrict__ out, float* __restrict__ tot) {
  const int blk = blockIdx.x;
  const int c   = blk % NCH;
  const int bh  = blk / NCH;
  const int b   = bh / NHEAD, h = bh % NHEAD;
  const int d   = threadIdx.x;
  float run = 0.0f;
  for (int r = 0; r < CHSZ; ++r) {
    const int l = c * CHSZ + r;
    run += vg[(((size_t)b * L_Q + l) * NHEAD + h) * DIM + d];
    out[((size_t)bh * L_Q + l) * DIM + d] = run;
  }
  tot[((size_t)bh * NCH + c) * DIM + d] = run;
}

__global__ __launch_bounds__(64) void cumsum_fix_kernel(
    float* __restrict__ out, const float* __restrict__ tot) {
  const int blk = blockIdx.x;
  const int c   = blk % NCH;
  const int bh  = blk / NCH;
  if (c == 0) return;
  const int d = threadIdx.x;
  float pre = 0.0f;
  for (int cc = 0; cc < c; ++cc) pre += tot[((size_t)bh * NCH + cc) * DIM + d];
  for (int r = 0; r < CHSZ; ++r) {
    const int l = c * CHSZ + r;
    out[((size_t)bh * L_Q + l) * DIM + d] += pre;
  }
}

// ---------------------------------------------------------------------------
// Kernel 4: flash-style sparse attention for the 40 selected queries of each
// (b,h). 3 waves = 3 M-tiles of 16 rows. K/V tiles are DMA'd by the Tensor
// Data Mover into LDS (f32), converted once to f16 working buffers, and fed to
// v_wmma_f32_16x16x32_f16. TDM for tile kb+1 overlaps the WMMA work on tile
// kb (single f32 staging buffer suffices: GEMMs read only the f16 copies).
// Online softmax with per-row causal cutoff at M_top[u]; overwrites the 40
// selected rows of d_out.
// ---------------------------------------------------------------------------
__global__ __launch_bounds__(96) void attn_topk_kernel(
    const float* __restrict__ qg, const float* __restrict__ kg,
    const float* __restrict__ vg, const int* __restrict__ mtop,
    float* __restrict__ out) {
  __shared__ int   Mt[MPAD];
  __shared__ float Kbuf[KBLK * DIM];  // TDM staging, f32 [key][d]
  __shared__ float Vbuf[KBLK * DIM];  // TDM staging, f32 [key][d]
  __shared__ f16   Qh[MPAD * DIM];    // selected queries, pre-scaled, f16
  __shared__ f16   Kh[KBLK * DIM];    // K block, f16 row-major [key][d]
  __shared__ f16   Vt[DIM * VSTR];    // V block, f16 transposed [d][key]
  __shared__ f16   Pl[MPAD * KBLK];   // softmax probs, row-major [u][key]

  const int bh  = blockIdx.x;
  const int b   = bh / NHEAD, h = bh % NHEAD;
  const int tid = threadIdx.x;
  const int lane = tid & 31;
  const int wv   = tid >> 5;         // wave id -> M-tile 0..2
  const int half = lane >> 4;        // C/D layout half
  const int N    = lane & 15;        // C/D layout column

  const float* kv_base = kg + ((size_t)b * L_Q * NHEAD + h) * DIM;
  const float* vv_base = vg + ((size_t)b * L_Q * NHEAD + h) * DIM;

  // Prologue: DMA tile 0 (wave 0 only; uniform branch skips waves 1-2).
  if (wv == 0) {
    tdm_load_tile_64x64_f32(kv_base, Kbuf);
    tdm_load_tile_64x64_f32(vv_base, Vbuf);
  }

  for (int u = tid; u < MPAD; u += 96)
    Mt[u] = (u < NTOP) ? mtop[bh * NTOP + u] : -1;
  __syncthreads();

  const float scale = 0.125f;  // 1/sqrt(64), folded into Q
  for (int i = tid; i < MPAD * DIM; i += 96) {
    const int u = i / DIM, d = i % DIM;
    float val = 0.0f;
    if (u < NTOP) {
      const int lq = Mt[u];
      val = qg[(((size_t)b * L_Q + lq) * NHEAD + h) * DIM + d] * scale;
    }
    Qh[i] = (f16)val;
  }

  float mrow[8], lrow[8];
  v8f acc[4];
#pragma unroll
  for (int vv = 0; vv < 8; ++vv) { mrow[vv] = NEGF; lrow[vv] = 0.0f; }
#pragma unroll
  for (int dt = 0; dt < 4; ++dt) acc[dt] = zero_v8f();

  for (int kb = 0; kb < NKB; ++kb) {
    // ---- wait for this tile's DMA, publish to all waves ----
    if (wv == 0) __builtin_amdgcn_s_wait_tensorcnt(0);
    __syncthreads();

    // ---- convert staged f32 -> f16 working buffers (K row-major, V^T) ----
    for (int i = tid; i < KBLK * DIM; i += 96) {
      const int key = i >> 6, d = i & 63;
      Kh[i] = (f16)Kbuf[i];
      Vt[d * VSTR + key] = (f16)Vbuf[i];
    }
    __syncthreads();  // staging buffers free from here on

    // ---- kick off DMA for the next tile; overlaps the WMMA work below ----
    if (wv == 0 && kb + 1 < NKB) {
      const size_t off = (size_t)(kb + 1) * KBLK * NHEAD * DIM;
      tdm_load_tile_64x64_f32(kv_base + off, Kbuf);
      tdm_load_tile_64x64_f32(vv_base + off, Vbuf);
    }

    // ---- S = (Q*scale) @ K^T for this wave's 16 rows x 64 keys ----
    v8f S[4];
    {
      const v16h a0 = load_A16(&Qh[(wv * 16) * DIM + 0], DIM);
      const v16h a1 = load_A16(&Qh[(wv * 16) * DIM + 32], DIM);
#pragma unroll
      for (int nt = 0; nt < 4; ++nt) {
        const v16h b0 = load_Bt16(&Kh[(nt * 16) * DIM + 0], DIM);
        const v16h b1 = load_Bt16(&Kh[(nt * 16) * DIM + 32], DIM);
        v8f c = zero_v8f();
        c = wmma_f16(a0, b0, c);
        c = wmma_f16(a1, b1, c);
        S[nt] = c;
      }
    }

    // ---- causal mask + online softmax (row stats per 16-lane half-group) ----
#pragma unroll
    for (int vv = 0; vv < 8; ++vv) {
      const int u   = wv * 16 + (half << 3) + vv;  // selected-query slot
      const int lim = Mt[u];                       // allow key col <= lim
      float rmax = NEGF;
#pragma unroll
      for (int nt = 0; nt < 4; ++nt) {
        const int col = kb * KBLK + nt * 16 + N;
        const float sv = (col <= lim) ? S[nt][vv] : NEGF;
        S[nt][vv] = sv;
        rmax = fmaxf(rmax, sv);
      }
#pragma unroll
      for (int off = 8; off >= 1; off >>= 1)
        rmax = fmaxf(rmax, __shfl_xor(rmax, off, 32));
      const float mnew  = fmaxf(mrow[vv], rmax);
      const bool  live  = (mnew > -1.0e38f);
      const float alpha = live ? __expf(mrow[vv] - mnew) : 1.0f;
      float rsum = 0.0f;
#pragma unroll
      for (int nt = 0; nt < 4; ++nt) {
        const float p = live ? __expf(S[nt][vv] - mnew) : 0.0f;
        S[nt][vv] = p;
        rsum += p;
      }
#pragma unroll
      for (int off = 8; off >= 1; off >>= 1)
        rsum += __shfl_xor(rsum, off, 32);
      lrow[vv] = lrow[vv] * alpha + rsum;
      mrow[vv] = mnew;
#pragma unroll
      for (int dt = 0; dt < 4; ++dt) acc[dt][vv] *= alpha;
#pragma unroll
      for (int nt = 0; nt < 4; ++nt)
        Pl[u * KBLK + nt * 16 + N] = (f16)S[nt][vv];
    }
    __syncthreads();

    // ---- acc += P @ V_blk  (K-dim = 64 keys, two k-steps of 32) ----
    {
      const v16h a0 = load_A16(&Pl[(wv * 16) * KBLK + 0], KBLK);
      const v16h a1 = load_A16(&Pl[(wv * 16) * KBLK + 32], KBLK);
#pragma unroll
      for (int dt = 0; dt < 4; ++dt) {
        const v16h b0 = load_Bt16(&Vt[(dt * 16) * VSTR + 0], VSTR);
        const v16h b1 = load_Bt16(&Vt[(dt * 16) * VSTR + 32], VSTR);
        acc[dt] = wmma_f16(a0, b0, acc[dt]);
        acc[dt] = wmma_f16(a1, b1, acc[dt]);
      }
    }
    __syncthreads();  // next iteration overwrites Kh/Vt
  }

  // ---- normalize and scatter into the selected output rows ----
#pragma unroll
  for (int vv = 0; vv < 8; ++vv) {
    const int u = wv * 16 + (half << 3) + vv;
    if (u < NTOP) {
      const float invl = (lrow[vv] > 0.0f) ? 1.0f / lrow[vv] : 0.0f;
      const int lq = Mt[u];
#pragma unroll
      for (int dt = 0; dt < 4; ++dt)
        out[((size_t)bh * L_Q + lq) * DIM + dt * 16 + N] = acc[dt][vv] * invl;
    }
  }
}

// ---------------------------------------------------------------------------
// Host-side launcher
// ---------------------------------------------------------------------------
extern "C" void kernel_launch(void* const* d_in, const int* in_sizes, int n_in,
                              void* d_out, int out_size, void* d_ws, size_t ws_size,
                              hipStream_t stream) {
  (void)in_sizes; (void)n_in; (void)out_size; (void)ws_size;
  const float*     qg   = (const float*)d_in[0];      // (B,L,H,D) f32
  const float*     kg   = (const float*)d_in[1];      // (B,L,H,D) f32
  const float*     vg   = (const float*)d_in[2];      // (B,L,H,D) f32
  const long long* idxs = (const long long*)d_in[3];  // (L, 40) i64
  // d_in[4] = attn_mask: unused by the reference.
  float* out = (float*)d_out;                         // (B,H,L,D) f32

  char* ws = (char*)d_ws;
  float* Mws  = (float*)ws;                                        // B*H*L f32
  int*   mtop = (int*)(ws + (size_t)NBAT * NHEAD * L_Q * sizeof(float));
  float* tot  = (float*)((char*)mtop + (size_t)NBAT * NHEAD * NTOP * sizeof(int));

  const int nwaves = NBAT * NHEAD * L_Q;
  sample_score_kernel<<<(nwaves * 32 + 255) / 256, 256, 0, stream>>>(qg, kg, idxs, Mws);
  topk_kernel<<<NBAT * NHEAD, 256, 0, stream>>>(Mws, mtop);
  cumsum_part_kernel<<<NBAT * NHEAD * NCH, 64, 0, stream>>>(vg, out, tot);
  cumsum_fix_kernel<<<NBAT * NHEAD * NCH, 64, 0, stream>>>(out, tot);
  attn_topk_kernel<<<NBAT * NHEAD, 96, 0, stream>>>(qg, kg, vg, mtop, out);
}
